// SPGAttentionSpatial_38044820308470
// MI455X (gfx1250) — matrix operand order repaired and checked by reading, not compiled, not run on hardware
//
#include <hip/hip_runtime.h>

// ---------------------------------------------------------------------------
// Problem constants
// ---------------------------------------------------------------------------
#define BATCH   4
#define CCH     256
#define HH      48
#define WW      48
#define SS      (HH * WW)          // 2304
#define NHEAD   8
#define DHEAD   32                 // CCH / NHEAD
#define GROUPS  16
#define CPG     (CCH / GROUPS)     // 16 channels per group
#define QKVROWS (3 * CCH)          // 768

typedef __attribute__((ext_vector_type(16))) _Float16 v16h;
typedef __attribute__((ext_vector_type(8)))  _Float16 v8h;
typedef __attribute__((ext_vector_type(8)))  float    v8f;

#define WMMA_F16(a, b, c) \
  __builtin_amdgcn_wmma_f32_16x16x32_f16(false, (a), false, (b), (short)0, (c), false, false)

// ---------------------------------------------------------------------------
// WMMA operand loaders (CDNA5 ISA 7.12.2 layouts, wave32)
// A 16x32 f16: lane m = lane&15; halfs 0..7 = K kb..kb+7, halfs 8..15 = K kb+16..kb+23,
//              kb = (lane>=16) ? 8 : 0.
// B 32x16 f16: lane n = lane&15; halfs 0..7 = K kb..kb+7, halfs 8..15 = K kb+16..kb+23, kb = (lane>=16) ? 8 : 0.
// Both reduce to contiguous 16-byte loads when K is the innermost memory dim.
// ---------------------------------------------------------------------------
__device__ __forceinline__ v16h load_A16(const _Float16* base, int ld, int lane) {
  int row = lane & 15;
  int kb  = (lane >> 4) << 3;                 // 0 or 8
  const _Float16* p = base + (size_t)row * ld + kb;
  union { v16h v; v8h h[2]; } u;
  u.h[0] = *(const v8h*)(p);
  u.h[1] = *(const v8h*)(p + 16);
  return u.v;
}

__device__ __forceinline__ v16h load_B16(const _Float16* base, int ld, int lane) {
  int col = lane & 15;
  int kb  = (lane >> 4) << 3;                 // 0 or 8
  const _Float16* p = base + (size_t)col * ld + kb;
  union { v16h v; v8h h[2]; } u;
  u.h[0] = *(const v8h*)(p);
  u.h[1] = *(const v8h*)(p + 16);
  return u.v;
}

// Cross-lane reductions within each 16-lane half (rows of the C-layout tile).
__device__ __forceinline__ float rmax16(float x) {
#pragma unroll
  for (int m = 1; m < 16; m <<= 1) x = fmaxf(x, __shfl_xor(x, m, 32));
  return x;
}
__device__ __forceinline__ float rsum16(float x) {
#pragma unroll
  for (int m = 1; m < 16; m <<= 1) x += __shfl_xor(x, m, 32);
  return x;
}

// ---------------------------------------------------------------------------
// K0: f32 -> f16 weight conversion
// ---------------------------------------------------------------------------
__global__ void cvt_f16_kernel(const float* __restrict__ src,
                               _Float16* __restrict__ dst, int n) {
  int i = blockIdx.x * blockDim.x + threadIdx.x;
  if (i < n) dst[i] = (_Float16)src[i];
}

// ---------------------------------------------------------------------------
// K1: GroupNorm. One block per (batch, group). Writes normed as (B, S, C) f16.
// ---------------------------------------------------------------------------
__global__ __launch_bounds__(256) void gn_kernel(
    const float* __restrict__ x, const float* __restrict__ w,
    const float* __restrict__ b, _Float16* __restrict__ normed) {
  int blk   = blockIdx.x;
  int batch = blk / GROUPS;
  int g     = blk % GROUPS;
  const float* xg = x + ((size_t)batch * CCH + (size_t)g * CPG) * SS;
  const int N = CPG * SS;                      // 36864

  float s = 0.f, s2 = 0.f;
  for (int i = threadIdx.x; i < N; i += blockDim.x) {
    float v = xg[i];
    s += v; s2 += v * v;
  }
  __shared__ float red[512];
  int tid = threadIdx.x;
  red[tid] = s; red[256 + tid] = s2;
  __syncthreads();
  for (int off = 128; off > 0; off >>= 1) {
    if (tid < off) { red[tid] += red[tid + off]; red[256 + tid] += red[256 + tid + off]; }
    __syncthreads();
  }
  float mean = red[0] / (float)N;
  float var  = red[256] / (float)N - mean * mean;
  float rstd = rsqrtf(var + 1e-5f);

  for (int i = threadIdx.x; i < N; i += blockDim.x) {
    int cl = i / SS, sp = i % SS;
    int c  = g * CPG + cl;
    float v = (xg[i] - mean) * rstd * w[c] + b[c];
    normed[((size_t)batch * SS + sp) * CCH + c] = (_Float16)v;
  }
}

// ---------------------------------------------------------------------------
// K2: QKV GEMM via WMMA. out[b,o,s] = sum_c wqkv[o,c] * normed[b,s,c].
// One wave -> 16x64 output (1 M-tile x 4 N-tiles), K loop 256 in steps of 32.
// Scatter q,k into (bh, S, d) f16; v into (bh, d, S) f16.
// ---------------------------------------------------------------------------
__global__ __launch_bounds__(256) void qkv_kernel(
    const _Float16* __restrict__ wq, const _Float16* __restrict__ normed,
    _Float16* __restrict__ qb, _Float16* __restrict__ kb,
    _Float16* __restrict__ vb) {
  const int MT = QKVROWS / 16;                 // 48
  const int NG = SS / 64;                      // 36 groups of 4 n-tiles
  int lane = threadIdx.x & 31;
  int wid  = (blockIdx.x * blockDim.x + threadIdx.x) >> 5;
  int batch = wid / (MT * NG);
  int rem   = wid % (MT * NG);
  int mt    = rem / NG;
  int ng    = rem % NG;

  const _Float16* nb = normed + (size_t)batch * SS * CCH;
  v8f acc[4] = {};

  for (int k0 = 0; k0 < CCH; k0 += 32) {
    v16h a = load_A16(wq + (size_t)mt * 16 * CCH + k0, CCH, lane);
#pragma unroll
    for (int j = 0; j < 4; ++j) {
      int n0 = (ng * 4 + j) * 16;
      v16h bm = load_B16(nb + (size_t)n0 * CCH + k0, CCH, lane);
      acc[j] = WMMA_F16(a, bm, acc[j]);
    }
  }

#pragma unroll
  for (int j = 0; j < 4; ++j) {
    int scol = (ng * 4 + j) * 16 + (lane & 15);
#pragma unroll
    for (int r = 0; r < 8; ++r) {
      int orow = mt * 16 + r + ((lane >> 4) << 3);
      int head = orow / 96;
      int trip = (orow % 96) / 32;
      int d    = orow % 32;
      size_t bh = (size_t)batch * NHEAD + head;
      _Float16 val = (_Float16)acc[j][r];
      if (trip == 0)      qb[(bh * SS + scol) * DHEAD + d] = val;
      else if (trip == 1) kb[(bh * SS + scol) * DHEAD + d] = val;
      else                vb[(bh * DHEAD + d) * SS + scol] = val;
    }
  }
}

// ---------------------------------------------------------------------------
// K3: Flash attention. One wave per (b, head, 16-row q-tile).
// Per 32-key chunk: 2 score WMMAs, online softmax (shfl-based row stats),
// p staged through LDS (C-layout -> A-layout transpose), 2 accumulate WMMAs.
// ---------------------------------------------------------------------------
__global__ __launch_bounds__(256) void attn_kernel(
    const _Float16* __restrict__ qb, const _Float16* __restrict__ kb,
    const _Float16* __restrict__ vb, _Float16* __restrict__ ob) {
  __shared__ _Float16 stage[8 * 16 * 32];      // 1 KB per wave
  int lane  = threadIdx.x & 31;
  int lwave = threadIdx.x >> 5;
  int wid   = (blockIdx.x * blockDim.x + threadIdx.x) >> 5;
  int bh = wid / (SS / 16);
  int qt = wid % (SS / 16);

  const _Float16* qp = qb + ((size_t)bh * SS + (size_t)qt * 16) * DHEAD;
  const _Float16* kp = kb + (size_t)bh * SS * DHEAD;
  const _Float16* vp = vb + (size_t)bh * DHEAD * SS;
  _Float16* sg = &stage[lwave * 512];

  // Pre-scale q by 1/sqrt(C) = 1/16 (exact in f16)
  v16h aq = load_A16(qp, DHEAD, lane);
  {
    const _Float16 sc = (_Float16)(1.0f / 16.0f);
#pragma unroll
    for (int i = 0; i < 16; ++i) aq[i] = aq[i] * sc;
  }

  float m[8], l[8];
#pragma unroll
  for (int r = 0; r < 8; ++r) { m[r] = -1e30f; l[r] = 0.f; }
  v8f acc0 = {}, acc1 = {};
  const v8f zero = {};

  for (int t0 = 0; t0 < SS; t0 += 32) {
    v16h bk0 = load_B16(kp + (size_t)t0 * DHEAD, DHEAD, lane);
    v16h bk1 = load_B16(kp + (size_t)(t0 + 16) * DHEAD, DHEAD, lane);
    v8f s0 = WMMA_F16(aq, bk0, zero);
    v8f s1 = WMMA_F16(aq, bk1, zero);

    int row = ((lane >> 4) << 3);
    int col = lane & 15;
#pragma unroll
    for (int r = 0; r < 8; ++r) {
      float tm = fmaxf(rmax16(s0[r]), rmax16(s1[r]));
      float mn = fmaxf(m[r], tm);
      float al = __expf(m[r] - mn);
      float p0 = __expf(s0[r] - mn);
      float p1 = __expf(s1[r] - mn);
      l[r] = l[r] * al + rsum16(p0) + rsum16(p1);
      m[r] = mn;
      acc0[r] *= al;
      acc1[r] *= al;
      sg[(row + r) * 32 + col]      = (_Float16)p0;
      sg[(row + r) * 32 + col + 16] = (_Float16)p1;
    }
    // Intra-wave LDS RAW: stores above feed the transposed A-layout read below.
    asm volatile("s_wait_dscnt 0x0" ::: "memory");

    v16h ap  = load_A16(sg, 32, lane);
    v16h bv0 = load_B16(vp + t0, SS, lane);                 // d cols 0..15
    v16h bv1 = load_B16(vp + (size_t)16 * SS + t0, SS, lane); // d cols 16..31
    acc0 = WMMA_F16(ap, bv0, acc0);
    acc1 = WMMA_F16(ap, bv1, acc1);
  }

  // Write o as (B, S, C) f16, c = head*32 + d
  int b = bh / NHEAD, h = bh % NHEAD;
  int d0 = lane & 15;
#pragma unroll
  for (int r = 0; r < 8; ++r) {
    int srow = qt * 16 + r + ((lane >> 4) << 3);
    float inv = 1.0f / l[r];
    size_t base = ((size_t)b * SS + srow) * CCH + (size_t)h * DHEAD;
    ob[base + d0]      = (_Float16)(acc0[r] * inv);
    ob[base + 16 + d0] = (_Float16)(acc1[r] * inv);
  }
}

// ---------------------------------------------------------------------------
// K4: Output projection + bias + residual.
// out[b,oc,s] = sum_c wout[oc,c]*o[b,s,c] + bias[oc] + input[b,oc,s]
// ---------------------------------------------------------------------------
__global__ __launch_bounds__(256) void outproj_kernel(
    const _Float16* __restrict__ wo, const _Float16* __restrict__ ob,
    const float* __restrict__ bias, const float* __restrict__ input,
    float* __restrict__ out) {
  const int MT = CCH / 16;                     // 16
  const int NG = SS / 64;                      // 36
  int lane = threadIdx.x & 31;
  int wid  = (blockIdx.x * blockDim.x + threadIdx.x) >> 5;
  int batch = wid / (MT * NG);
  int rem   = wid % (MT * NG);
  int mt    = rem / NG;
  int ng    = rem % NG;

  const _Float16* obb = ob + (size_t)batch * SS * CCH;
  v8f acc[4] = {};

  for (int k0 = 0; k0 < CCH; k0 += 32) {
    v16h a = load_A16(wo + (size_t)mt * 16 * CCH + k0, CCH, lane);
#pragma unroll
    for (int j = 0; j < 4; ++j) {
      int n0 = (ng * 4 + j) * 16;
      v16h bm = load_B16(obb + (size_t)n0 * CCH + k0, CCH, lane);
      acc[j] = WMMA_F16(a, bm, acc[j]);
    }
  }

#pragma unroll
  for (int j = 0; j < 4; ++j) {
    int sc = (ng * 4 + j) * 16 + (lane & 15);
#pragma unroll
    for (int r = 0; r < 8; ++r) {
      int oc = mt * 16 + r + ((lane >> 4) << 3);
      size_t idx = ((size_t)batch * CCH + oc) * SS + sc;
      out[idx] = acc[j][r] + bias[oc] + input[idx];
    }
  }
}

// ---------------------------------------------------------------------------
// Host launcher
// ---------------------------------------------------------------------------
extern "C" void kernel_launch(void* const* d_in, const int* in_sizes, int n_in,
                              void* d_out, int out_size, void* d_ws, size_t ws_size,
                              hipStream_t stream) {
  const float* input = (const float*)d_in[0];
  const float* gnw   = (const float*)d_in[1];
  const float* gnb   = (const float*)d_in[2];
  const float* wqkv  = (const float*)d_in[3];
  const float* wout  = (const float*)d_in[4];
  const float* bout  = (const float*)d_in[5];
  float* out = (float*)d_out;

  char* ws = (char*)d_ws;
  size_t off = 0;
  auto wsalloc = [&](size_t bytes) {
    void* p = ws + off;
    off = (off + bytes + 255) & ~(size_t)255;
    return p;
  };
  _Float16* wqkv_h = (_Float16*)wsalloc((size_t)QKVROWS * CCH * 2);
  _Float16* wout_h = (_Float16*)wsalloc((size_t)CCH * CCH * 2);
  _Float16* normed = (_Float16*)wsalloc((size_t)BATCH * SS * CCH * 2);
  _Float16* qb     = (_Float16*)wsalloc((size_t)BATCH * NHEAD * SS * DHEAD * 2);
  _Float16* kb     = (_Float16*)wsalloc((size_t)BATCH * NHEAD * SS * DHEAD * 2);
  _Float16* vb     = (_Float16*)wsalloc((size_t)BATCH * NHEAD * SS * DHEAD * 2);
  _Float16* ob     = (_Float16*)wsalloc((size_t)BATCH * SS * CCH * 2);

  cvt_f16_kernel<<<(QKVROWS * CCH + 255) / 256, 256, 0, stream>>>(wqkv, wqkv_h, QKVROWS * CCH);
  cvt_f16_kernel<<<(CCH * CCH + 255) / 256, 256, 0, stream>>>(wout, wout_h, CCH * CCH);

  gn_kernel<<<BATCH * GROUPS, 256, 0, stream>>>(input, gnw, gnb, normed);

  // 4 * 48 * 36 = 6912 waves -> 864 blocks of 8 waves
  qkv_kernel<<<864, 256, 0, stream>>>(wqkv_h, normed, qb, kb, vb);

  // 4 * 8 * 144 = 4608 waves -> 576 blocks
  attn_kernel<<<576, 256, 0, stream>>>(qb, kb, vb, ob);

  // 4 * 16 * 36 = 2304 waves -> 288 blocks
  outproj_kernel<<<288, 256, 0, stream>>>(wout_h, ob, bout, input, out);
}
